// MultiHeadAttention_65764539236424
// MI455X (gfx1250) — matrix operand order repaired and checked by reading, not compiled, not run on hardware
//
#include <hip/hip_runtime.h>

#define D_MODEL   1024
#define NUM_HEADS 16
#define DEPTH     64
#define BATCH     2
#define SEQ       2048

typedef __attribute__((ext_vector_type(16))) __bf16 v16bf;
typedef __attribute__((ext_vector_type(8)))  float  v8f;
typedef unsigned int v4u __attribute__((ext_vector_type(4)));
typedef int          v8i_ __attribute__((ext_vector_type(8)));
typedef int          v4i_ __attribute__((ext_vector_type(4)));

union Frag { uint4 u[2]; v16bf v; };
union U4   { uint4 u; unsigned short s[8]; };

__device__ __forceinline__ unsigned short f2bf(float f) {
  unsigned int u = __float_as_uint(f);
  u += 0x7FFFu + ((u >> 16) & 1u);          // round-to-nearest-even
  return (unsigned short)(u >> 16);
}

__device__ __forceinline__ v8f wmma_bf16(const Frag& a, const Frag& b, v8f c) {
  return __builtin_amdgcn_wmma_f32_16x16x32_bf16(false, a.v, false, b.v,
                                                 (short)0, c, false, false);
}

// ---------------------------------------------------------------------------
// CDNA5 async global->LDS staging (ASYNCcnt), with sync fallback
// ---------------------------------------------------------------------------
#if defined(__has_builtin)
#if __has_builtin(__builtin_amdgcn_global_load_async_to_lds_b128)
#define HAVE_ASYNC_LDS 1
#endif
#if __has_builtin(__builtin_amdgcn_tensor_load_to_lds) && \
    __has_builtin(__builtin_amdgcn_s_wait_tensorcnt)
#define HAVE_TDM 1
#endif
#endif
#ifndef HAVE_ASYNC_LDS
#define HAVE_ASYNC_LDS 0
#endif
#ifndef HAVE_TDM
#define HAVE_TDM 0
#endif

__device__ __forceinline__ void async_copy16(const void* g, void* l) {
#if HAVE_ASYNC_LDS
  // param 1: addrspace(1) int4*  (source in global)
  // param 2: addrspace(3) int4*  (destination in LDS)
  __builtin_amdgcn_global_load_async_to_lds_b128(
      (__attribute__((address_space(1))) v4i_*)(void*)(uintptr_t)g,
      (__attribute__((address_space(3))) v4i_*)l, 0, 0);
#else
  *(uint4*)l = *(const uint4*)g;
#endif
}

template <int N>
__device__ __forceinline__ void wait_async() {
#if HAVE_ASYNC_LDS
#if __has_builtin(__builtin_amdgcn_s_wait_asynccnt)
  __builtin_amdgcn_s_wait_asynccnt(N);
#else
  asm volatile("s_wait_asynccnt %0" ::"i"(N) : "memory");
#endif
#endif
}

// ---------------------------------------------------------------------------
// f32 -> bf16 conversion (vectorized x4)
// ---------------------------------------------------------------------------
__global__ __launch_bounds__(256) void cvt_f32_bf16(const float* __restrict__ x,
                                                    unsigned short* __restrict__ y,
                                                    int n4) {
  int i = blockIdx.x * blockDim.x + threadIdx.x;
  if (i < n4) {
    float4 f = ((const float4*)x)[i];
    ushort4 r;
    r.x = f2bf(f.x); r.y = f2bf(f.y); r.z = f2bf(f.z); r.w = f2bf(f.w);
    ((ushort4*)y)[i] = r;
  }
}

// ---------------------------------------------------------------------------
// NT GEMM:  out[m,n] = sum_k A[m,k]*W[n,k] + bias[n]
//   mode 0: f32 row-major [M][N]          mode 1: bf16 split-heads [B,H,S,64]
// Block tile 256x64, 8 waves x (32 rows x 64 cols), K step 32.
// Double-buffered LDS, async global->LDS fill of tile t+1 overlapped with
// WMMAs on tile t; each thread issues exactly 5 async b128 copies per tile,
// so `s_wait_asynccnt 5` at the loop head retires the *previous* tile
// (async loads complete in order, ISA 08 §4.1).
// ---------------------------------------------------------------------------
#define TM 256
#define TN 64
#define TK 32
#define LDT 40   // LDS row stride (elems): 32 + 8 pad -> 80B rows, 16B aligned

__global__ __launch_bounds__(256) void gemm_bf16_nt(
    const unsigned short* __restrict__ A,   // [M][K] bf16
    const unsigned short* __restrict__ W,   // [N][K] bf16
    const float* __restrict__ bias,         // [N]
    float* __restrict__ outF,
    unsigned short* __restrict__ outH,
    int M, int N, int K, int mode) {
  __shared__ unsigned short sA[2][TM * LDT];   // 2 x 20480 B
  __shared__ unsigned short sB[2][TN * LDT];   // 2 x  5120 B

  const int t    = threadIdx.x;
  const int lane = t & 31;
  const int w    = t >> 5;
  const int lm   = lane & 15;
  const int half = lane >> 4;

  const int n0 = blockIdx.x * TN;
  const int m0 = blockIdx.y * TM;

  v8f acc[2][4];
#pragma unroll
  for (int r = 0; r < 2; ++r)
#pragma unroll
    for (int i = 0; i < 4; ++i) acc[r][i] = (v8f){0,0,0,0,0,0,0,0};

  // 5 x b128 per thread per tile: A = 1024 copies (ids 0..1023), B = 256.
  auto stage = [&](int buf, int k0) {
#pragma unroll
    for (int j = 0; j < 5; ++j) {
      const int L = t + 256 * j;
      if (L < 1024) {
        const int row = L >> 2, col = (L & 3) * 8;
        async_copy16(A + (size_t)(m0 + row) * K + k0 + col,
                     sA[buf] + row * LDT + col);
#if !HAVE_ASYNC_LDS
        if (k0 + TK < K)
          __builtin_prefetch(A + (size_t)(m0 + row) * K + k0 + TK + col, 0, 1);
#endif
      } else {
        const int L2 = L - 1024;
        const int row = L2 >> 2, col = (L2 & 3) * 8;
        async_copy16(W + (size_t)(n0 + row) * K + k0 + col,
                     sB[buf] + row * LDT + col);
      }
    }
  };

  const int nTiles = K / TK;
  stage(0, 0);
  for (int kt = 0; kt < nTiles; ++kt) {
    const int cur = kt & 1;
    if (kt + 1 < nTiles) {
      stage(cur ^ 1, (kt + 1) * TK);
      wait_async<5>();          // previous tile's copies have landed
    } else {
      wait_async<0>();
    }
    __syncthreads();            // tile `cur` visible to all waves

    const unsigned short* sAb = sA[cur];
    const unsigned short* sBb = sB[cur];

    Frag af[2];                 // two 16x32 A fragments -> 32 rows per wave
#pragma unroll
    for (int r = 0; r < 2; ++r) {
      const unsigned short* ap = sAb + (w * 32 + r * 16 + lm) * LDT + 8 * half;
      af[r].u[0] = *(const uint4*)(ap);
      af[r].u[1] = *(const uint4*)(ap + 16);
    }
#pragma unroll
    for (int nt = 0; nt < 4; ++nt) {
      Frag bf;                  // 32x16 B fragment, reused by both A frags
      const unsigned short* bp = sBb + (nt * 16 + lm) * LDT + 8 * half;
      bf.u[0] = *(const uint4*)(bp);
      bf.u[1] = *(const uint4*)(bp + 16);
      acc[0][nt] = wmma_bf16(af[0], bf, acc[0][nt]);
      acc[1][nt] = wmma_bf16(af[1], bf, acc[1][nt]);
    }
    __syncthreads();            // protect buffer reuse two tiles out
  }

  // ---- epilogue: bias + store --------------------------------------------
#pragma unroll
  for (int r = 0; r < 2; ++r) {
    const int mw = m0 + w * 32 + r * 16;
#pragma unroll
    for (int nt = 0; nt < 4; ++nt) {
      const int col = n0 + nt * 16 + lm;
      const float bv = bias[col];
#pragma unroll
      for (int i = 0; i < 8; ++i) {
        const int row = mw + i + 8 * half;
        const float val = acc[r][nt][i] + bv;
        if (mode == 0) {
          outF[(size_t)row * N + col] = val;
        } else {
          const int b = row >> 11;
          const int s = row & (SEQ - 1);
          const int h = col >> 6;
          const int d = col & (DEPTH - 1);
          outH[(((size_t)(b * NUM_HEADS + h)) * SEQ + s) * DEPTH + d] = f2bf(val);
        }
      }
    }
  }
}

// ---------------------------------------------------------------------------
// Fused flash attention.  Grid: (SEQ/64, H, B), 128 threads (4 waves).
// K chunk staged by the Tensor Data Mover when available (one tensor_load per
// 32x64 tile, D# pad fields reproduce the LDK=72 bank-padded layout).
// ---------------------------------------------------------------------------
#define CH  32
#define LDK 72   // 144B rows = 128B data + 16B pad
#define LDV 40
#define LDP 40

__global__ __launch_bounds__(128) void attention_fused(
    const unsigned short* __restrict__ qh,  // [B,H,S,DEPTH] bf16
    const unsigned short* __restrict__ kh,
    const unsigned short* __restrict__ vh,
    const float* __restrict__ mask,         // [B,1,S,S] f32
    unsigned short* __restrict__ attn_out)  // [B,S,D_MODEL] bf16
{
  __shared__ unsigned short sK[CH * LDK];
  __shared__ unsigned short sVt[DEPTH * LDV];
  __shared__ unsigned short sP[4][16 * LDP];

  const int t    = threadIdx.x;
  const int lane = t & 31;
  const int w    = t >> 5;
  const int lm   = lane & 15;
  const int half = lane >> 4;

  const int b  = blockIdx.z;
  const int h  = blockIdx.y;
  const int q0 = blockIdx.x * 64 + w * 16;

  const size_t head_off = ((size_t)(b * NUM_HEADS + h)) * SEQ * DEPTH;
  const unsigned short* qbase = qh + head_off;
  const unsigned short* kbase = kh + head_off;
  const unsigned short* vbase = vh + head_off;
  const float* mbase = mask + (size_t)b * SEQ * SEQ;

  Frag qa[2];
  {
    const unsigned short* qrow = qbase + (size_t)(q0 + lm) * DEPTH;
#pragma unroll
    for (int f = 0; f < 2; ++f) {
      qa[f].u[0] = *(const uint4*)(qrow + 32 * f + 8 * half);
      qa[f].u[1] = *(const uint4*)(qrow + 32 * f + 8 * half + 16);
    }
  }

  float mi[8], li[8];
  v8f o[4];
#pragma unroll
  for (int i = 0; i < 8; ++i) { mi[i] = -3.0e38f; li[i] = 0.0f; }
#pragma unroll
  for (int nt = 0; nt < 4; ++nt) o[nt] = (v8f){0,0,0,0,0,0,0,0};

  const float sscale = 0.125f;  // 1/sqrt(DEPTH)

#if HAVE_TDM
  const unsigned ldsK =
      (unsigned)(uintptr_t)(__attribute__((address_space(3))) void*)sK;
#endif

  for (int k0 = 0; k0 < SEQ; k0 += CH) {
    __syncthreads();
#if HAVE_TDM
    if (w == 0) {
      // D# for a 2D bf16 tile: tensor [SEQ-k0 rows x 64], tile [32 x 64],
      // row pitch 64, LDS pad: 4 DWORDs after every 32 DWORDs (-> LDK=72).
      const unsigned long long ga =
          (unsigned long long)(uintptr_t)(kbase + (size_t)k0 * DEPTH);
      v4u g0 = { 1u,                                   // count=1, user mode
                 ldsK,                                 // lds_addr
                 (unsigned)ga,                         // global_addr lo
                 (unsigned)(ga >> 32) | (2u << 30) };  // addr hi | type=2
      v8i_ g1 = { (int)((1u << 16)      // data_size = 2B
                      | (1u << 20)      // pad_enable
                      | (4u << 22)      // pad_interval: 32 DWORDs
                      | (3u << 25)),    // pad_amount:   4 DWORDs
                  (int)(64u << 16),     // tensor_dim0 = 64 (bits 79:48)
                  (int)((unsigned)(SEQ - k0) << 16),  // tensor_dim1 (111:80)
                  (int)(64u << 16),     // dim1 hi=0 | tile_dim0 = 64 (127:112)
                  (int)32,              // tile_dim1 = 32, tile_dim2 = 0
                  (int)64,              // tensor_dim0_stride = 64 (207:160)
                  0, 0 };
      v4i_ g2 = {0, 0, 0, 0};
      v4i_ g3 = {0, 0, 0, 0};
#if defined(__clang_major__) && (__clang_major__ <= 22)
      __builtin_amdgcn_tensor_load_to_lds(g0, g1, g2, g3, 0);
#else
      v8i_ g4 = {0, 0, 0, 0, 0, 0, 0, 0};
      __builtin_amdgcn_tensor_load_to_lds(g0, g1, g2, g3, g4, 0);
#endif
      __builtin_amdgcn_s_wait_tensorcnt(0);
    }
    // V chunk: manual transposed staging (TDM cannot transpose into [d][key])
    for (int it = t; it < 256; it += 128) {
      const int key = it >> 3;
      const int d0  = (it & 7) << 3;
      U4 vv;
      vv.u = *(const uint4*)(vbase + (size_t)(k0 + key) * DEPTH + d0);
#pragma unroll
      for (int j = 0; j < 8; ++j) sVt[(d0 + j) * LDV + key] = vv.s[j];
    }
#else
    for (int it = t; it < 256; it += 128) {
      const int key = it >> 3;
      const int d0  = (it & 7) << 3;
      *(uint4*)(sK + key * LDK + d0) =
          *(const uint4*)(kbase + (size_t)(k0 + key) * DEPTH + d0);
      U4 vv;
      vv.u = *(const uint4*)(vbase + (size_t)(k0 + key) * DEPTH + d0);
#pragma unroll
      for (int j = 0; j < 8; ++j) sVt[(d0 + j) * LDV + key] = vv.s[j];
    }
#endif
    __syncthreads();

    // ---- scores S = Q K^T ------------------------------------------------
    v8f s0 = (v8f){0,0,0,0,0,0,0,0};
    v8f s1 = (v8f){0,0,0,0,0,0,0,0};
#pragma unroll
    for (int f = 0; f < 2; ++f) {
      Frag bk0, bk1;  // B(k=d, n=key) = Ktile[key][d]
      const unsigned short* kp0 = sK + lm * LDK + 32 * f + 8 * half;
      const unsigned short* kp1 = sK + (16 + lm) * LDK + 32 * f + 8 * half;
      bk0.u[0] = *(const uint4*)(kp0); bk0.u[1] = *(const uint4*)(kp0 + 16);
      bk1.u[0] = *(const uint4*)(kp1); bk1.u[1] = *(const uint4*)(kp1 + 16);
      s0 = wmma_bf16(qa[f], bk0, s0);
      s1 = wmma_bf16(qa[f], bk1, s1);
    }

    // ---- online softmax --------------------------------------------------
    unsigned short* pw = sP[w];
#pragma unroll
    for (int i = 0; i < 8; ++i) {
      const int row = i + 8 * half;
      const float* mrow = mbase + (size_t)(q0 + row) * SEQ + k0;
      float v0 = s0[i] * sscale + mrow[lm]      * -1.0e9f;
      float v1 = s1[i] * sscale + mrow[lm + 16] * -1.0e9f;
      float cm = fmaxf(v0, v1);
#pragma unroll
      for (int off = 1; off < 16; off <<= 1)
        cm = fmaxf(cm, __shfl_xor(cm, off, 32));
      const float mn    = fmaxf(mi[i], cm);
      const float alpha = __expf(mi[i] - mn);
      const float p0    = __expf(v0 - mn);
      const float p1    = __expf(v1 - mn);
      float rs = p0 + p1;
#pragma unroll
      for (int off = 1; off < 16; off <<= 1)
        rs += __shfl_xor(rs, off, 32);
      li[i] = li[i] * alpha + rs;
      mi[i] = mn;
      o[0][i] *= alpha; o[1][i] *= alpha; o[2][i] *= alpha; o[3][i] *= alpha;
      pw[row * LDP + lm]      = f2bf(p0);
      pw[row * LDP + lm + 16] = f2bf(p1);
    }
    // same-wave DS ops are in-order: safe to re-read P in A layout

    // ---- O += P @ V ------------------------------------------------------
    Frag pf;
    const unsigned short* pp = pw + lm * LDP + 8 * half;
    pf.u[0] = *(const uint4*)(pp);
    pf.u[1] = *(const uint4*)(pp + 16);
#pragma unroll
    for (int nt = 0; nt < 4; ++nt) {
      Frag bv;  // B(k=key, n=d) = Vt[d][key]
      const unsigned short* vp = sVt + (nt * 16 + lm) * LDV + 8 * half;
      bv.u[0] = *(const uint4*)(vp);
      bv.u[1] = *(const uint4*)(vp + 16);
      o[nt] = wmma_bf16(pf, bv, o[nt]);
    }
  }

  // ---- normalize and store merged heads ----------------------------------
#pragma unroll
  for (int i = 0; i < 8; ++i) {
    const float inv = 1.0f / li[i];
    const int row = q0 + i + 8 * half;
    unsigned short* orow =
        attn_out + ((size_t)b * SEQ + row) * D_MODEL + h * DEPTH;
#pragma unroll
    for (int nt = 0; nt < 4; ++nt)
      orow[nt * 16 + lm] = f2bf(o[nt][i] * inv);
  }
}

// ---------------------------------------------------------------------------
// Host-side orchestration
// ---------------------------------------------------------------------------
extern "C" void kernel_launch(void* const* d_in, const int* in_sizes, int n_in,
                              void* d_out, int out_size, void* d_ws, size_t ws_size,
                              hipStream_t stream) {
  (void)in_sizes; (void)n_in; (void)out_size; (void)ws_size;
  const float* v    = (const float*)d_in[0];
  const float* k    = (const float*)d_in[1];
  const float* q    = (const float*)d_in[2];
  const float* mask = (const float*)d_in[3];
  const float* wq   = (const float*)d_in[4];
  const float* bq   = (const float*)d_in[5];
  const float* wk   = (const float*)d_in[6];
  const float* bk   = (const float*)d_in[7];
  const float* wv   = (const float*)d_in[8];
  const float* bv   = (const float*)d_in[9];
  const float* wo   = (const float*)d_in[10];
  const float* bo   = (const float*)d_in[11];

  char* ws = (char*)d_ws;
  const size_t NT = (size_t)BATCH * SEQ * D_MODEL;  // 4M elems
  const size_t NW = (size_t)D_MODEL * D_MODEL;      // 1M elems
  size_t off = 0;
  auto take = [&](size_t elems) {
    unsigned short* p = (unsigned short*)(ws + off);
    off += ((elems * 2 + 255) & ~(size_t)255);
    return p;
  };
  unsigned short* qb   = take(NT);
  unsigned short* kb   = take(NT);
  unsigned short* vb   = take(NT);
  unsigned short* wqb  = take(NW);
  unsigned short* wkb  = take(NW);
  unsigned short* wvb  = take(NW);
  unsigned short* wob  = take(NW);
  unsigned short* qhb  = take(NT);
  unsigned short* khb  = take(NT);
  unsigned short* vhb  = take(NT);
  unsigned short* attb = take(NT);  // ~64 MB of d_ws total

  const int n4t = (int)(NT / 4), n4w = (int)(NW / 4);
  cvt_f32_bf16<<<dim3((n4t + 255) / 256), 256, 0, stream>>>(q, qb, n4t);
  cvt_f32_bf16<<<dim3((n4t + 255) / 256), 256, 0, stream>>>(k, kb, n4t);
  cvt_f32_bf16<<<dim3((n4t + 255) / 256), 256, 0, stream>>>(v, vb, n4t);
  cvt_f32_bf16<<<dim3((n4w + 255) / 256), 256, 0, stream>>>(wq, wqb, n4w);
  cvt_f32_bf16<<<dim3((n4w + 255) / 256), 256, 0, stream>>>(wk, wkb, n4w);
  cvt_f32_bf16<<<dim3((n4w + 255) / 256), 256, 0, stream>>>(wv, wvb, n4w);
  cvt_f32_bf16<<<dim3((n4w + 255) / 256), 256, 0, stream>>>(wo, wob, n4w);

  const int M = BATCH * SEQ;
  dim3 gemmGrid(D_MODEL / TN, M / TM);
  gemm_bf16_nt<<<gemmGrid, 256, 0, stream>>>(qb, wqb, bq, nullptr, qhb,
                                             M, D_MODEL, D_MODEL, 1);
  gemm_bf16_nt<<<gemmGrid, 256, 0, stream>>>(kb, wkb, bk, nullptr, khb,
                                             M, D_MODEL, D_MODEL, 1);
  gemm_bf16_nt<<<gemmGrid, 256, 0, stream>>>(vb, wvb, bv, nullptr, vhb,
                                             M, D_MODEL, D_MODEL, 1);

  dim3 attnGrid(SEQ / 64, NUM_HEADS, BATCH);
  attention_fused<<<attnGrid, 128, 0, stream>>>(qhb, khb, vhb, mask, attb);

  gemm_bf16_nt<<<gemmGrid, 256, 0, stream>>>(attb, wob, bo, (float*)d_out,
                                             nullptr, M, D_MODEL, D_MODEL, 0);
}
